// Lnlv_71519795413647
// MI455X (gfx1250) — matrix operand order repaired and checked by reading, not compiled, not run on hardware
//
#include <hip/hip_runtime.h>
#include <math.h>

typedef __bf16 bf16_t;
typedef bf16_t v16bf __attribute__((ext_vector_type(16)));
typedef bf16_t v8bf  __attribute__((ext_vector_type(8)));
typedef float  v8f   __attribute__((ext_vector_type(8)));

#define HIDN 512
#define TN   512
#define SN   32
#define WEDN 300
#define WEDP 320
#define NWIN 497

__device__ __forceinline__ float sigf(float x) { return 1.0f / (1.0f + expf(-x)); }

// ---------------------------------------------------------------------------
// Weight prep: Wt[n*Kp + k] = (bf16)(k<K ? W[k*N + n] : 0).  Tiled LDS
// transpose so both the fp32 read and the bf16 write are coalesced.
// grid = (ceil(Kp/32), ceil(N/32)), block = (32,8).
// ---------------------------------------------------------------------------
__global__ void transpose_bf16(const float* __restrict__ W, int K, int N, int Kp,
                               bf16_t* __restrict__ Wt)
{
    __shared__ float tile[32][33];
    int kb = blockIdx.x * 32;
    int nb = blockIdx.y * 32;
    int tx = threadIdx.x;   // 0..31
    int ty = threadIdx.y;   // 0..7
    for (int i = ty; i < 32; i += 8) {
        int k = kb + i, n = nb + tx;
        tile[i][tx] = (k < K && n < N) ? W[(size_t)k * N + n] : 0.f;
    }
    __syncthreads();
    for (int i = ty; i < 32; i += 8) {
        int n = nb + i, k = kb + tx;
        if (n < N && k < Kp) Wt[(size_t)n * Kp + k] = (bf16_t)tile[tx][i];
    }
}

// ---------------------------------------------------------------------------
// GEMM: C[M,N] = act(A[M,K] @ W + bias), A fp32 row-major, W pre-transposed
// bf16 (N x Kp).  block = 128 (4 waves); wave w owns tile
// (m0 = by*16, n0 = (bx*4+w)*16).  Kp multiple of 32, M of 16, N of 64 —
// no guards, so fragment loads are straight-line vector loads:
//   A: 4 x b128 fp32 (+ packed cvt), B: 2 x b128 bf16 per K-step.
// act: 0 = none, 1 = tanh.
// ---------------------------------------------------------------------------
__global__ void wmma_gemm_bias(const float* __restrict__ A, int lda,
                               const bf16_t* __restrict__ Wt, int Kp,
                               const float* __restrict__ bias,
                               float* __restrict__ C, int ldc,
                               int act)
{
    int wave  = threadIdx.x >> 5;
    int lane  = threadIdx.x & 31;
    int n0    = (blockIdx.x * 4 + wave) * 16;
    int m0    = blockIdx.y * 16;
    int lhalf = (lane >> 4) & 1;
    int mrow  = m0 + (lane & 15);
    int ncol  = n0 + (lane & 15);

    // A fragment (ISA 7.12.2, 16-bit A 16x32): per lane, two contiguous
    // 8-element runs at k = lhalf*8 and k = 16 + lhalf*8.
    const float*  arow = A  + (size_t)mrow * lda + lhalf * 8;
    // B fragment: per lane, two contiguous 8-element runs at k = lhalf*8 and k = 16 + lhalf*8.
    const bf16_t* brow = Wt + (size_t)ncol * Kp  + lhalf * 8;

    v8f c = {0.f, 0.f, 0.f, 0.f, 0.f, 0.f, 0.f, 0.f};

    for (int kk = 0; kk < Kp; kk += 32) {
        float4 p0 = *(const float4*)(arow + kk);
        float4 p1 = *(const float4*)(arow + kk + 4);
        float4 p2 = *(const float4*)(arow + kk + 16);
        float4 p3 = *(const float4*)(arow + kk + 20);
        v16bf a;
        a[0]  = (bf16_t)p0.x; a[1]  = (bf16_t)p0.y;
        a[2]  = (bf16_t)p0.z; a[3]  = (bf16_t)p0.w;
        a[4]  = (bf16_t)p1.x; a[5]  = (bf16_t)p1.y;
        a[6]  = (bf16_t)p1.z; a[7]  = (bf16_t)p1.w;
        a[8]  = (bf16_t)p2.x; a[9]  = (bf16_t)p2.y;
        a[10] = (bf16_t)p2.z; a[11] = (bf16_t)p2.w;
        a[12] = (bf16_t)p3.x; a[13] = (bf16_t)p3.y;
        a[14] = (bf16_t)p3.z; a[15] = (bf16_t)p3.w;

        v8bf b0 = *(const v8bf*)(brow + kk);
        v8bf b1 = *(const v8bf*)(brow + kk + 16);
        v16bf b = __builtin_shufflevector(b0, b1, 0, 1, 2, 3, 4, 5, 6, 7,
                                          8, 9, 10, 11, 12, 13, 14, 15);

        c = __builtin_amdgcn_wmma_f32_16x16x32_bf16(false, a, false, b,
                                                    (short)0, c, false, false);
    }

    float bv = bias[ncol];
#pragma unroll
    for (int i = 0; i < 8; ++i) {              // D: M = (lane/16)*8 + i
        int m = m0 + lhalf * 8 + i;
        float o = c[i] + bv;
        if (act == 1) o = tanhf(o);
        C[(size_t)m * ldc + ncol] = o;
    }
}

// ---------------------------------------------------------------------------
// Embedding gather into K-padded layout: Et[s, 0:300] = emb[text[s]], rest 0.
// ---------------------------------------------------------------------------
__global__ void gather_emb(const float* __restrict__ emb,
                           const int* __restrict__ text,
                           float* __restrict__ out)
{
    int s = blockIdx.x;
    int row = text[s];
    for (int w = threadIdx.x; w < WEDP; w += blockDim.x)
        out[s * WEDP + w] = (w < WEDN) ? emb[(size_t)row * WEDN + w] : 0.f;
}

// ---------------------------------------------------------------------------
// Persistent-workgroup GRU recurrence. grid = 1 (uni) or 2 (bi, dir=blockIdx.x,
// dir 1 scans reversed). h lives in LDS; Whh streams from L2 each step
// (coalesced along N) with prefetch. pre = x@Wih+bih precomputed by GEMM.
// ---------------------------------------------------------------------------
__global__ void gru_seq(const float* __restrict__ pre0,
                        const float* __restrict__ pre1,
                        const float* __restrict__ Whh,  // [dir][H][3H]
                        const float* __restrict__ bhh,  // [dir][3H]
                        float* __restrict__ Y,          // [T][ldy], col off dir*H
                        int T, int H, int ldy)
{
    __shared__ float h_lds[HIDN];
    __shared__ float ph_lds[3 * HIDN];

    int dir = blockIdx.x;
    const float* pre = dir ? pre1 : pre0;
    const float* Wd  = Whh + (size_t)dir * H * 3 * H;
    const float* bd  = bhh + (size_t)dir * 3 * H;
    int NG = 3 * H;
    int nt = blockDim.x;

    for (int j = threadIdx.x; j < H; j += nt) h_lds[j] = 0.f;
    __syncthreads();

    for (int step = 0; step < T; ++step) {
        int t = dir ? (T - 1 - step) : step;

        for (int n = threadIdx.x; n < NG; n += nt) {
            float acc = bd[n];
            const float* wp = Wd + n;
            for (int k = 0; k < H; ++k) {
                if ((k & 31) == 0 && (k + 32) < H)
                    __builtin_prefetch(wp + (size_t)(k + 32) * NG, 0, 1);
                acc = fmaf(h_lds[k], wp[(size_t)k * NG], acc);
            }
            ph_lds[n] = acc;
        }
        __syncthreads();

        const float* pt = pre + (size_t)t * NG;
        for (int j = threadIdx.x; j < H; j += nt) {
            float r  = sigf(pt[j] + ph_lds[j]);
            float z  = sigf(pt[H + j] + ph_lds[H + j]);
            float nn = tanhf(pt[2 * H + j] + r * ph_lds[2 * H + j]);
            float hn = (1.f - z) * nn + z * h_lds[j];
            h_lds[j] = hn;
            Y[(size_t)t * ldy + dir * H + j] = hn;
        }
        __syncthreads();
    }
}

// ---------------------------------------------------------------------------
// Cross-modal attention, fused per t; writes concat [h_v_t | h_s_t].
// ---------------------------------------------------------------------------
__global__ void cma_fuse(const float* __restrict__ Q,    // T x HID
                         const float* __restrict__ Kk,   // S x HID
                         const float* __restrict__ vvec, // HID
                         const float* __restrict__ Hv,   // T x HID
                         const float* __restrict__ Hs,   // S x HID
                         float* __restrict__ cat1)       // T x 2*HID
{
    __shared__ float q[HIDN];
    __shared__ float sc[SN];
    int t = blockIdx.x;
    int tid = threadIdx.x, lane = tid & 31, wid = tid >> 5;

    for (int h = tid; h < HIDN; h += blockDim.x) q[h] = Q[(size_t)t * HIDN + h];
    __syncthreads();

    for (int s = wid; s < SN; s += 8) {
        float acc = 0.f;
        const float* kr = Kk + (size_t)s * HIDN;
        for (int h = lane; h < HIDN; h += 32)
            acc += tanhf(q[h] + kr[h]) * vvec[h];
        for (int m = 16; m >= 1; m >>= 1) acc += __shfl_xor(acc, m, 32);
        if (lane == 0) sc[s] = acc;
    }
    __syncthreads();

    if (tid < 32) {
        float v = sc[tid];
        float mx = v;
        for (int m = 16; m >= 1; m >>= 1) mx = fmaxf(mx, __shfl_xor(mx, m, 32));
        float e = expf(v - mx);
        float sm = e;
        for (int m = 16; m >= 1; m >>= 1) sm += __shfl_xor(sm, m, 32);
        sc[tid] = e / sm;
    }
    __syncthreads();

    for (int h = tid; h < HIDN; h += blockDim.x) {
        float hb = 0.f;
        for (int s = 0; s < SN; ++s) hb += sc[s] * Hs[(size_t)s * HIDN + h];
        float hv  = Hv[(size_t)t * HIDN + h];
        float hvt = fmaxf(hv, 0.f) * hb;
        float hst = fmaxf(hb, 0.f) * hvt;
        cat1[(size_t)t * 2 * HIDN + h]        = hvt;
        cat1[(size_t)t * 2 * HIDN + HIDN + h] = hst;
    }
}

// ---------------------------------------------------------------------------
// Self-interaction attention: only the unmasked u>=t region is computed
// (mask = -1e30 elsewhere => exact). Writes concat [h_r | att].
// ---------------------------------------------------------------------------
__global__ void self_attn(const float* __restrict__ Q2,
                          const float* __restrict__ K2,
                          const float* __restrict__ vvec,
                          const float* __restrict__ hr,
                          float* __restrict__ cat2)      // T x 2*HID
{
    __shared__ float q[HIDN];
    __shared__ float p[TN];
    __shared__ float red[8];
    int t = blockIdx.x;
    int tid = threadIdx.x, lane = tid & 31, wid = tid >> 5;

    for (int h = tid; h < HIDN; h += blockDim.x) q[h] = Q2[(size_t)t * HIDN + h];
    __syncthreads();

    for (int u = t + wid; u < TN; u += 8) {
        float acc = 0.f;
        const float* kr = K2 + (size_t)u * HIDN;
        for (int h = lane; h < HIDN; h += 32)
            acc += tanhf(q[h] + kr[h]) * vvec[h];
        for (int m = 16; m >= 1; m >>= 1) acc += __shfl_xor(acc, m, 32);
        if (lane == 0) p[u] = acc;
    }
    __syncthreads();

    float lmax = -3.4e38f;
    for (int u = t + tid; u < TN; u += blockDim.x) lmax = fmaxf(lmax, p[u]);
    for (int m = 16; m >= 1; m >>= 1) lmax = fmaxf(lmax, __shfl_xor(lmax, m, 32));
    if (lane == 0) red[wid] = lmax;
    __syncthreads();
    float bmax = red[0];
    for (int i = 1; i < 8; ++i) bmax = fmaxf(bmax, red[i]);
    __syncthreads();

    float lsum = 0.f;
    for (int u = t + tid; u < TN; u += blockDim.x) {
        float e = expf(p[u] - bmax);
        p[u] = e;
        lsum += e;
    }
    for (int m = 16; m >= 1; m >>= 1) lsum += __shfl_xor(lsum, m, 32);
    if (lane == 0) red[wid] = lsum;
    __syncthreads();
    float bsum = 0.f;
    for (int i = 0; i < 8; ++i) bsum += red[i];
    float inv = 1.f / bsum;
    __syncthreads();
    for (int u = t + tid; u < TN; u += blockDim.x) p[u] *= inv;
    __syncthreads();

    for (int h = tid; h < HIDN; h += blockDim.x) {
        float acc = 0.f;
        for (int u = t; u < TN; ++u) acc += p[u] * hr[(size_t)u * HIDN + h];
        cat2[(size_t)t * 2 * HIDN + HIDN + h] = acc;
        cat2[(size_t)t * 2 * HIDN + h]        = hr[(size_t)t * HIDN + h];
    }
}

// h_o = sum_s H_s  (exact: softmax over a length-1 axis == 1 => wp_* dead)
__global__ void colsum(const float* __restrict__ Hs, float* __restrict__ ho)
{
    int h = blockIdx.x * blockDim.x + threadIdx.x;
    if (h < HIDN) {
        float a = 0.f;
        for (int s = 0; s < SN; ++s) a += Hs[(size_t)s * HIDN + h];
        ho[h] = a;
    }
}

__global__ void build_cat3(const float* __restrict__ hd,
                           const float* __restrict__ ho,
                           float* __restrict__ cat3)
{
    int t = blockIdx.x;
    for (int h = threadIdx.x; h < HIDN; h += blockDim.x) {
        cat3[(size_t)t * 2 * HIDN + h]        = hd[(size_t)t * HIDN + h];
        cat3[(size_t)t * 2 * HIDN + HIDN + h] = ho[h];
    }
}

__global__ void final_score(const float* __restrict__ tmp,
                            const float* __restrict__ cpv,
                            float* __restrict__ out)
{
    __shared__ float red[4];
    int t = blockIdx.x;
    int tid = threadIdx.x, lane = tid & 31, wid = tid >> 5;
    float acc = 0.f;
    for (int h = tid; h < HIDN; h += blockDim.x) acc += tmp[(size_t)t * HIDN + h] * cpv[h];
    for (int m = 16; m >= 1; m >>= 1) acc += __shfl_xor(acc, m, 32);
    if (lane == 0) red[wid] = acc;
    __syncthreads();
    if (tid == 0 && t < NWIN) {
        out[t] = red[0] + red[1] + red[2] + red[3];
        ((int*)out)[NWIN + t] = t;
    }
}

// ---------------------------------------------------------------------------
extern "C" void kernel_launch(void* const* d_in, const int* in_sizes, int n_in,
                              void* d_out, int out_size, void* d_ws, size_t ws_size,
                              hipStream_t stream)
{
    (void)in_sizes; (void)n_in; (void)out_size; (void)ws_size;

    const float* video     = (const float*)d_in[0];
    const int*   text      = (const int*)  d_in[1];
    const float* vp_W      = (const float*)d_in[2];
    const float* vp_b      = (const float*)d_in[3];
    const float* vgru_Wih  = (const float*)d_in[4];
    const float* vgru_Whh  = (const float*)d_in[5];
    const float* vgru_bih  = (const float*)d_in[6];
    const float* vgru_bhh  = (const float*)d_in[7];
    const float* emb       = (const float*)d_in[8];
    const float* tp_W      = (const float*)d_in[9];
    const float* tp_b      = (const float*)d_in[10];
    const float* tgru_Wih  = (const float*)d_in[11];
    const float* tgru_Whh  = (const float*)d_in[12];
    const float* tgru_bih  = (const float*)d_in[13];
    const float* tgru_bhh  = (const float*)d_in[14];
    const float* cma_Wq    = (const float*)d_in[15];
    const float* cma_bq    = (const float*)d_in[16];
    const float* cma_Wk    = (const float*)d_in[17];
    const float* cma_bk    = (const float*)d_in[18];
    const float* cma_v     = (const float*)d_in[19];
    const float* cm_Wih    = (const float*)d_in[20];
    const float* cm_Whh    = (const float*)d_in[21];
    const float* cm_bih    = (const float*)d_in[22];
    const float* cm_bhh    = (const float*)d_in[23];
    const float* si_Wq     = (const float*)d_in[24];
    const float* si_bq     = (const float*)d_in[25];
    const float* si_Wk     = (const float*)d_in[26];
    const float* si_bk     = (const float*)d_in[27];
    const float* si_v      = (const float*)d_in[28];
    const float* sig_Wih   = (const float*)d_in[29];
    const float* sig_Whh   = (const float*)d_in[30];
    const float* sig_bih   = (const float*)d_in[31];
    const float* sig_bhh   = (const float*)d_in[32];
    // d_in[33..35] = wp_* : provably dead (softmax over length-1 axis == 1)
    const float* cp_W1     = (const float*)d_in[36];
    const float* cp_b1     = (const float*)d_in[37];
    const float* cp_v      = (const float*)d_in[38];

    // ---- fp32 workspace carve-out ----
    float* ws   = (float*)d_ws;
    float* Xv   = ws;                       // 512*512
    float* preA = Xv   + 512 * 512;         // 512*1536
    float* preB = preA + 512 * 1536;        // 512*768
    float* Hv   = preB + 512 * 768;         // 512*512
    float* Etp  = Hv   + 512 * 512;         // 32*320 (K-padded)
    float* Xt   = Etp  + 32 * WEDP;         // 32*512
    float* Hs   = Xt   + 32 * 512;          // 32*512
    float* Qb   = Hs   + 32 * 512;          // 512*512
    float* Kb   = Qb   + 512 * 512;         // 32*512
    float* K2   = Kb   + 32 * 512;          // 512*512
    float* cat  = K2   + 512 * 512;         // 512*1024
    float* hr   = cat  + 512 * 1024;        // 512*512
    float* hd   = hr   + 512 * 512;         // 512*512
    float* ho   = hd   + 512 * 512;         // 512
    float* tmp  = ho   + 512;               // 512*512

    // ---- bf16 transposed-weight carve-out ----
    bf16_t* bws    = (bf16_t*)(tmp + 512 * 512);
    bf16_t* vpWt   = bws;                   // 512  x 1024
    bf16_t* vIh0   = vpWt + 512 * 1024;     // 768  x 512
    bf16_t* vIh1   = vIh0 + 768 * 512;      // 768  x 512
    bf16_t* tpWt   = vIh1 + 768 * 512;      // 512  x 320
    bf16_t* tIh0   = tpWt + 512 * WEDP;     // 768  x 512
    bf16_t* tIh1   = tIh0 + 768 * 512;      // 768  x 512
    bf16_t* cmaQt  = tIh1 + 768 * 512;      // 512  x 512
    bf16_t* cmaKt  = cmaQt + 512 * 512;     // 512  x 512
    bf16_t* cmIt   = cmaKt + 512 * 512;     // 1536 x 1024
    bf16_t* siQt   = cmIt + 1536 * 1024;    // 512  x 512
    bf16_t* siKt   = siQt + 512 * 512;      // 512  x 512
    bf16_t* sigIt  = siKt + 512 * 512;      // 1536 x 1024
    bf16_t* cpW1t  = sigIt + 1536 * 1024;   // 512  x 1024

    dim3 tb(32, 8);
    // K rows, N cols, Kp padded
    transpose_bf16<<<dim3(32, 16), tb, 0, stream>>>(vp_W, 1024, 512, 1024, vpWt);
    transpose_bf16<<<dim3(16, 24), tb, 0, stream>>>(vgru_Wih,             512, 768, 512, vIh0);
    transpose_bf16<<<dim3(16, 24), tb, 0, stream>>>(vgru_Wih + 512 * 768, 512, 768, 512, vIh1);
    transpose_bf16<<<dim3(10, 16), tb, 0, stream>>>(tp_W, 300, 512, WEDP, tpWt);
    transpose_bf16<<<dim3(16, 24), tb, 0, stream>>>(tgru_Wih,             512, 768, 512, tIh0);
    transpose_bf16<<<dim3(16, 24), tb, 0, stream>>>(tgru_Wih + 512 * 768, 512, 768, 512, tIh1);
    transpose_bf16<<<dim3(16, 16), tb, 0, stream>>>(cma_Wq, 512, 512, 512, cmaQt);
    transpose_bf16<<<dim3(16, 16), tb, 0, stream>>>(cma_Wk, 512, 512, 512, cmaKt);
    transpose_bf16<<<dim3(32, 48), tb, 0, stream>>>(cm_Wih, 1024, 1536, 1024, cmIt);
    transpose_bf16<<<dim3(16, 16), tb, 0, stream>>>(si_Wq, 512, 512, 512, siQt);
    transpose_bf16<<<dim3(16, 16), tb, 0, stream>>>(si_Wk, 512, 512, 512, siKt);
    transpose_bf16<<<dim3(32, 48), tb, 0, stream>>>(sig_Wih, 1024, 1536, 1024, sigIt);
    transpose_bf16<<<dim3(32, 16), tb, 0, stream>>>(cp_W1, 1024, 512, 1024, cpW1t);

    // ---- video path ----
    wmma_gemm_bias<<<dim3(8, 32),  128, 0, stream>>>(video, 1024, vpWt, 1024, vp_b, Xv, 512, 0);
    wmma_gemm_bias<<<dim3(12, 32), 128, 0, stream>>>(Xv, 512, vIh0, 512, vgru_bih,       preA, 768, 0);
    wmma_gemm_bias<<<dim3(12, 32), 128, 0, stream>>>(Xv, 512, vIh1, 512, vgru_bih + 768, preB, 768, 0);
    gru_seq<<<2, 1024, 0, stream>>>(preA, preB, vgru_Whh, vgru_bhh, Hv, 512, 256, 512);

    // ---- text path ----
    gather_emb<<<32, 128, 0, stream>>>(emb, text, Etp);
    wmma_gemm_bias<<<dim3(8, 2),  128, 0, stream>>>(Etp, WEDP, tpWt, WEDP, tp_b, Xt, 512, 0);
    wmma_gemm_bias<<<dim3(12, 2), 128, 0, stream>>>(Xt, 512, tIh0, 512, tgru_bih,       preA, 768, 0);
    wmma_gemm_bias<<<dim3(12, 2), 128, 0, stream>>>(Xt, 512, tIh1, 512, tgru_bih + 768, preB, 768, 0);
    gru_seq<<<2, 1024, 0, stream>>>(preA, preB, tgru_Whh, tgru_bhh, Hs, 32, 256, 512);

    // ---- cross-modal attention + co-memory GRU ----
    wmma_gemm_bias<<<dim3(8, 32), 128, 0, stream>>>(Hv, 512, cmaQt, 512, cma_bq, Qb, 512, 0);
    wmma_gemm_bias<<<dim3(8, 2),  128, 0, stream>>>(Hs, 512, cmaKt, 512, cma_bk, Kb, 512, 0);
    cma_fuse<<<512, 256, 0, stream>>>(Qb, Kb, cma_v, Hv, Hs, cat);
    wmma_gemm_bias<<<dim3(24, 32), 128, 0, stream>>>(cat, 1024, cmIt, 1024, cm_bih, preA, 1536, 0);
    gru_seq<<<1, 1024, 0, stream>>>(preA, preA, cm_Whh, cm_bhh, hr, 512, 512, 512);

    // ---- self-interaction attention + GRU ----
    wmma_gemm_bias<<<dim3(8, 32), 128, 0, stream>>>(hr, 512, siQt, 512, si_bq, Qb, 512, 0);
    wmma_gemm_bias<<<dim3(8, 32), 128, 0, stream>>>(hr, 512, siKt, 512, si_bk, K2, 512, 0);
    self_attn<<<512, 256, 0, stream>>>(Qb, K2, si_v, hr, cat);
    wmma_gemm_bias<<<dim3(24, 32), 128, 0, stream>>>(cat, 1024, sigIt, 1024, sig_bih, preA, 1536, 0);
    gru_seq<<<1, 1024, 0, stream>>>(preA, preA, sig_Whh, sig_bhh, hd, 512, 512, 512);

    // ---- output head ----
    colsum<<<2, 256, 0, stream>>>(Hs, ho);
    build_cat3<<<512, 256, 0, stream>>>(hd, ho, cat);
    wmma_gemm_bias<<<dim3(8, 32), 128, 0, stream>>>(cat, 1024, cpW1t, 1024, cp_b1, tmp, 512, 1);
    final_score<<<512, 128, 0, stream>>>(tmp, cp_v, (float*)d_out);
}
